// EncoderBlock_23021024707099
// MI455X (gfx1250) — matrix operand order repaired and checked by reading, not compiled, not run on hardware
//
#include <hip/hip_runtime.h>
#include <hip/hip_bf16.h>
#include <math.h>

// Problem constants (match reference)
#define Bv 32
#define Tv 2048
#define Ev 512
#define Fv 512
#define KC 7
#define Hh 8
#define HD 64
#define EPSv 1e-5f
#define Mtot (Bv * Tv)          // 65536 rows for the flat GEMMs
#define PPART 64                // partial blocks per batch for LN stats

typedef __attribute__((ext_vector_type(16))) __bf16 v16bf;
typedef __attribute__((ext_vector_type(8)))  __bf16 v8bf;
typedef __attribute__((ext_vector_type(8)))  float  v8f;
typedef unsigned short ushort_t;
typedef unsigned int   uint_t;
typedef uint_t __attribute__((ext_vector_type(4))) v4u;
typedef int    __attribute__((ext_vector_type(8))) v8i;
typedef int    __attribute__((ext_vector_type(4))) v4i;

// ---------------- bf16 <-> f32 helpers (bit-level, RNE) ----------------
__device__ __forceinline__ ushort_t f2bf(float f) {
    uint_t u = __float_as_uint(f);
    u += 0x7fffu + ((u >> 16) & 1u);
    return (ushort_t)(u >> 16);
}
__device__ __forceinline__ float bf2f(ushort_t h) {
    return __uint_as_float(((uint_t)h) << 16);
}

// Positional encoding quirk: pe indexed by *batch* position b, broadcast over T.
__device__ __forceinline__ float pe_val(int b, int e) {
    int i = e >> 1;
    float div = __expf((float)i * -0.035977892f);   // -2*ln(10000)/512
    float a = (float)b * div;
    return (e & 1) ? __cosf(a) : __sinf(a);
}

// ---------------- CDNA5 async copy: global -> LDS (ASYNCcnt tracked) ----------------
__device__ __forceinline__ void async_b128(uint_t lds_byte, const ushort_t* g) {
    asm volatile("global_load_async_to_lds_b128 %0, %1, off"
                 :: "v"(lds_byte), "v"(g) : "memory");
}
__device__ __forceinline__ void wait_async_0() { asm volatile("s_wait_asynccnt 0x0" ::: "memory"); }

// ---------------- CDNA5 Tensor Data Mover: one instruction stages a whole tile ----------------
// 2D tensor (rows x 512 bf16), tile = 128 rows x 64 halves, LDS-padded +8 halves/row
// (pad_interval = 32 DWORDs -> code 4, pad_amount = 4 DWORDs -> code 3) => LDS stride 72.
__device__ __forceinline__ void tdm_load_tile(const ushort_t* gsrc, uint_t lds_byte) {
    unsigned long long ga = (unsigned long long)(uintptr_t)gsrc;
    v4u g0;
    g0.x = 1u;                                   // count=1, user descriptor
    g0.y = lds_byte;                             // lds_addr
    g0.z = (uint_t)(ga & 0xffffffffu);           // global_addr[31:0]
    g0.w = (uint_t)((ga >> 32) & 0x01ffffffu)    // global_addr[56:32]
         | 0x80000000u;                          // type=2 (bits 127:126 = 10b)
    v8i g1;
    g1[0] = (int)0x07110000u;   // wg_mask=0 | data_size=2B | pad_enable | interval=32dw | amount=4dw
    g1[1] = (int)(512u << 16);  // tensor_dim0 = 512 (bits 79:48, low16 here)
    g1[2] = 0;                  // dim0 hi | tensor_dim1 low16 (65536 -> low16 = 0)
    g1[3] = (int)(1u | (64u << 16));   // tensor_dim1 hi16 = 1 | tile_dim0 = 64
    g1[4] = (int)128u;          // tile_dim1 = 128 | tile_dim2 = 0
    g1[5] = (int)512u;          // tensor_dim0_stride = 512 (low32)
    g1[6] = 0;                  // stride hi | tensor_dim1_stride low16
    g1[7] = 0;
    v4i gz = {0, 0, 0, 0};
#if defined(__clang_major__) && (__clang_major__ >= 23)
    v8i gz8 = {0, 0, 0, 0, 0, 0, 0, 0};
    __builtin_amdgcn_tensor_load_to_lds(g0, g1, gz, gz, gz8, 0);
#else
    __builtin_amdgcn_tensor_load_to_lds(g0, g1, gz, gz, 0);
#endif
}

// ---------------- fp32 -> bf16 weight conversion ----------------
__global__ void k_f2bf(const float* __restrict__ in, ushort_t* __restrict__ out, int n) {
    int i = blockIdx.x * blockDim.x + threadIdx.x;
    if (i < n) out[i] = f2bf(in[i]);
}

// ---------------- two-stage LN stats, per-b over (T,C) ----------------
__global__ void k_stats_pe_part(const float* __restrict__ x, float* __restrict__ part) {
    int b = blockIdx.y, p = blockIdx.x;
    const float* xb = x + (size_t)b * Tv * Ev;
    const int n = (Tv * Ev) / PPART;
    int base = p * n;
    float s = 0.f, s2 = 0.f;
    for (int i = threadIdx.x; i < n; i += blockDim.x) {
        int idx = base + i;
        float v = xb[idx] + pe_val(b, idx & (Ev - 1));
        s += v; s2 += v * v;
    }
    __shared__ float r1[256], r2[256];
    r1[threadIdx.x] = s; r2[threadIdx.x] = s2;
    __syncthreads();
    for (int o = 128; o > 0; o >>= 1) {
        if (threadIdx.x < o) { r1[threadIdx.x] += r1[threadIdx.x + o]; r2[threadIdx.x] += r2[threadIdx.x + o]; }
        __syncthreads();
    }
    if (threadIdx.x == 0) { part[(b * PPART + p) * 2] = r1[0]; part[(b * PPART + p) * 2 + 1] = r2[0]; }
}

__global__ void k_stats_b_part(const float* __restrict__ x, float* __restrict__ part) {
    int b = blockIdx.y, p = blockIdx.x;
    const float* xb = x + (size_t)b * Tv * Fv;
    const int n = (Tv * Fv) / PPART;
    int base = p * n;
    float s = 0.f, s2 = 0.f;
    for (int i = threadIdx.x; i < n; i += blockDim.x) {
        float v = xb[base + i];
        s += v; s2 += v * v;
    }
    __shared__ float r1[256], r2[256];
    r1[threadIdx.x] = s; r2[threadIdx.x] = s2;
    __syncthreads();
    for (int o = 128; o > 0; o >>= 1) {
        if (threadIdx.x < o) { r1[threadIdx.x] += r1[threadIdx.x + o]; r2[threadIdx.x] += r2[threadIdx.x + o]; }
        __syncthreads();
    }
    if (threadIdx.x == 0) { part[(b * PPART + p) * 2] = r1[0]; part[(b * PPART + p) * 2 + 1] = r2[0]; }
}

__global__ void k_stats_fin(const float* __restrict__ part, float* __restrict__ stats, float invn) {
    int b = blockIdx.x;
    if (threadIdx.x == 0) {
        float s = 0.f, s2 = 0.f;
        for (int p = 0; p < PPART; p++) {
            s  += part[(b * PPART + p) * 2];
            s2 += part[(b * PPART + p) * 2 + 1];
        }
        float mu  = s * invn;
        float var = s2 * invn - mu * mu;
        stats[b * 2]     = mu;
        stats[b * 2 + 1] = rsqrtf(var + EPSv);
    }
}

// ---------------- per-t LN stats over (B,F) (post-attention LN) ----------------
__global__ void k_stats_t(const float* __restrict__ x, float* __restrict__ stats) {
    int t = blockIdx.x;
    float s = 0.f, s2 = 0.f;
    for (int i = threadIdx.x; i < Bv * Fv; i += blockDim.x) {
        int b = i >> 9, f = i & (Fv - 1);
        float v = x[(size_t)b * Tv * Fv + (size_t)t * Fv + f];
        s += v; s2 += v * v;
    }
    __shared__ float r1[256], r2[256];
    r1[threadIdx.x] = s; r2[threadIdx.x] = s2;
    __syncthreads();
    for (int o = 128; o > 0; o >>= 1) {
        if (threadIdx.x < o) { r1[threadIdx.x] += r1[threadIdx.x + o]; r2[threadIdx.x] += r2[threadIdx.x + o]; }
        __syncthreads();
    }
    if (threadIdx.x == 0) {
        float inv = 1.0f / (float)(Bv * Fv);
        float mu  = r1[0] * inv;
        float var = r2[0] * inv - mu * mu;
        stats[t * 2]     = mu;
        stats[t * 2 + 1] = rsqrtf(var + EPSv);
    }
}

// ---------------- plain normalize -> bf16 (feeds WMMA GEMMs) ----------------
__global__ void k_norm_b(const float* __restrict__ x, const float* __restrict__ stats,
                         ushort_t* __restrict__ out) {
    int i = blockIdx.x * blockDim.x + threadIdx.x;
    if (i >= Mtot * Fv) return;
    int b = i >> 20;
    float mu = stats[b * 2], rs = stats[b * 2 + 1];
    out[i] = f2bf((x[i] - mu) * rs);
}

__global__ void k_norm_t(const float* __restrict__ x, const float* __restrict__ stats,
                         ushort_t* __restrict__ out) {
    int i = blockIdx.x * blockDim.x + threadIdx.x;
    if (i >= Mtot * Fv) return;
    int t = (i >> 9) & (Tv - 1);
    float mu = stats[t * 2], rs = stats[t * 2 + 1];
    out[i] = f2bf((x[i] - mu) * rs);
}

// ---------------- fused LayerNorm + depthwise conv (K=7 along T) ----------------
__global__ void k_ln_dwconv_pe(const float* __restrict__ x, const float* __restrict__ stats,
                               const float* __restrict__ w, const float* __restrict__ bias,
                               ushort_t* __restrict__ out) {
    int i = blockIdx.x * blockDim.x + threadIdx.x;
    if (i >= Mtot * Ev) return;
    int c  = i & (Ev - 1);
    int t  = (i >> 9) & (Tv - 1);
    int b  = i >> 20;
    float mu = stats[b * 2], rs = stats[b * 2 + 1];
    float pe = pe_val(b, c);
    float acc = bias[c];
#pragma unroll
    for (int k = 0; k < KC; k++) {
        int tt = t + k - KC / 2;
        if (tt >= 0 && tt < Tv)
            acc += w[c * KC + k] * ((x[i + (k - KC / 2) * Ev] + pe - mu) * rs);
    }
    out[i] = f2bf(acc);
}

__global__ void k_ln_dwconv(const float* __restrict__ x, const float* __restrict__ stats,
                            const float* __restrict__ w, const float* __restrict__ bias,
                            ushort_t* __restrict__ out) {
    int i = blockIdx.x * blockDim.x + threadIdx.x;
    if (i >= Mtot * Fv) return;
    int c  = i & (Fv - 1);
    int t  = (i >> 9) & (Tv - 1);
    int b  = i >> 20;
    float mu = stats[b * 2], rs = stats[b * 2 + 1];
    float acc = bias[c];
#pragma unroll
    for (int k = 0; k < KC; k++) {
        int tt = t + k - KC / 2;
        if (tt >= 0 && tt < Tv)
            acc += w[c * KC + k] * ((x[i + (k - KC / 2) * Fv] - mu) * rs);
    }
    out[i] = f2bf(acc);
}

// ---------------- WMMA GEMM with TDM-staged, double-buffered A tiles ----------------
// out[m,n] = sum_k A[m,k]*W[n,k] + bias[n] (+res).  A: Mtot x 512 bf16, W: 512 x 512 bf16.
// Block: 8 wave32 waves, tile 128(M) x 64(N), K tile 64. Wave 0 drives the TDM engine.
template<int RES, int OUTBF>
__global__ __launch_bounds__(256) void k_gemm(
    const ushort_t* __restrict__ A, const ushort_t* __restrict__ W,
    const float* __restrict__ bias, const float* res,
    float* outf, ushort_t* outb)
{
    const int LDA2 = 72;                       // TDM pad gives 64 + 8 halves per row
    __shared__ __align__(16) ushort_t As[2][128 * LDA2];

    int tid  = threadIdx.x;
    int lane = tid & 31, wave = tid >> 5;
    int wm = wave & 3, wn = wave >> 2;         // 4x2 wave grid: each wave -> 32x32 output
    int r0 = blockIdx.y * 128;
    int n0 = blockIdx.x * 64;
    int hs   = lane >> 4;                      // half-wave select (ISA 16-bit K interleave)
    int lrow = lane & 15;

    v8f zero = {0.f, 0.f, 0.f, 0.f, 0.f, 0.f, 0.f, 0.f};
    v8f acc[2][2];
    acc[0][0] = zero; acc[0][1] = zero; acc[1][0] = zero; acc[1][1] = zero;

    const ushort_t* gtile = A + (size_t)r0 * 512;
    uint_t lds0 = (uint_t)(uintptr_t)&As[0][0];
    uint_t lds1 = (uint_t)(uintptr_t)&As[1][0];

    if (wave == 0) tdm_load_tile(gtile, lds0);
    int buf = 0;
    for (int io = 0; io < 8; io++) {
        int ko = io * 64;
        if (wave == 0) {
            if (io < 7) {
                tdm_load_tile(gtile + ko + 64, buf ? lds0 : lds1);
                __builtin_amdgcn_s_wait_tensorcnt(1);   // current tile complete
            } else {
                __builtin_amdgcn_s_wait_tensorcnt(0);
            }
        }
        __syncthreads();                       // tile visible to all waves

#pragma unroll
        for (int ks = 0; ks < 64; ks += 32) {
            v16bf afrag[2], bfrag[2];
            // A fragment (16x32): lane(0-15)=row M; elems 0..7 -> K=hs*8+j, 8..15 -> K=16+hs*8+j
#pragma unroll
            for (int mi = 0; mi < 2; mi++) {
                const ushort_t* ap = &As[buf][(wm * 32 + mi * 16 + lrow) * LDA2 + ks];
                v8bf lo = *(const v8bf*)(ap + hs * 8);
                v8bf hi = *(const v8bf*)(ap + 16 + hs * 8);
                afrag[mi] = __builtin_shufflevector(lo, hi, 0, 1, 2, 3, 4, 5, 6, 7,
                                                    8, 9, 10, 11, 12, 13, 14, 15);
            }
            // B fragment (32x16): lane(0-15)=col N; elems 0..7 -> K=hs*8+j, 8..15 -> K=16+hs*8+j
#pragma unroll
            for (int ni = 0; ni < 2; ni++) {
                const ushort_t* wrow =
                    W + (size_t)(n0 + wn * 32 + ni * 16 + lrow) * 512 + ko + ks;
                v8bf lo = *(const v8bf*)(wrow + hs * 8);
                v8bf hi = *(const v8bf*)(wrow + 16 + hs * 8);
                bfrag[ni] = __builtin_shufflevector(lo, hi, 0, 1, 2, 3, 4, 5, 6, 7,
                                                    8, 9, 10, 11, 12, 13, 14, 15);
            }
#pragma unroll
            for (int mi = 0; mi < 2; mi++)
#pragma unroll
                for (int ni = 0; ni < 2; ni++)
                    acc[mi][ni] = __builtin_amdgcn_wmma_f32_16x16x32_bf16(
                        false, afrag[mi], false, bfrag[ni], (short)0, acc[mi][ni], false, false);
        }
        __syncthreads();                       // done reading this buffer before reuse
        buf ^= 1;
    }

    // epilogue: C/D layout -> lane: N = lane&15, M = i + 8*(lane>>4)
#pragma unroll
    for (int mi = 0; mi < 2; mi++) {
#pragma unroll
        for (int ni = 0; ni < 2; ni++) {
            int n = n0 + wn * 32 + ni * 16 + lrow;
            float bn = bias[n];
#pragma unroll
            for (int i = 0; i < 8; i++) {
                int m = r0 + wm * 32 + mi * 16 + hs * 8 + i;
                size_t off = (size_t)m * 512 + n;
                float v = acc[mi][ni][i] + bn;
                if (RES) v += res[off];
                if (OUTBF) outb[off] = f2bf(v);
                else       outf[off] = v;
            }
        }
    }
}

// ---------------- batch-axis attention: per (t,h), 32x32 scores over hd=64 ----------------
// K/V rows staged to LDS via async-to-LDS (shared by all lanes); Q row lives in registers.
__global__ void k_attn(const ushort_t* __restrict__ q, const ushort_t* __restrict__ k,
                       const ushort_t* __restrict__ v, ushort_t* __restrict__ o) {
    int t = blockIdx.x;
    int h = blockIdx.y;
    int b = threadIdx.x;                 // one lane per batch row (wave32)
    __shared__ __align__(16) ushort_t Kh[32 * HD];
    __shared__ __align__(16) ushort_t Vh[32 * HD];
    __shared__ float Ss[32][33];

    size_t base = (size_t)b * Tv * Fv + (size_t)t * Fv + (size_t)h * HD;
    uint_t lk = (uint_t)(uintptr_t)&Kh[b * HD];
    uint_t lv = (uint_t)(uintptr_t)&Vh[b * HD];
#pragma unroll
    for (int j = 0; j < HD / 8; j++) {          // 8 halves (16B) per async op
        async_b128(lk + j * 16, k + base + j * 8);
        async_b128(lv + j * 16, v + base + j * 8);
    }
    float qr[HD];
    {
        const uint_t* qp = (const uint_t*)(q + base);
#pragma unroll
        for (int j = 0; j < HD / 2; j++) {
            uint_t u = qp[j];
            qr[2 * j]     = bf2f((ushort_t)u);
            qr[2 * j + 1] = bf2f((ushort_t)(u >> 16));
        }
    }
    wait_async_0();
    __syncthreads();

    float mx = -1e30f;
    for (int c = 0; c < 32; c++) {
        float d = 0.f;
        const uint_t* kr = (const uint_t*)&Kh[c * HD];
#pragma unroll
        for (int j = 0; j < HD / 2; j++) {
            uint_t u = kr[j];                    // broadcast LDS read
            d += qr[2 * j] * bf2f((ushort_t)u);
            d += qr[2 * j + 1] * bf2f((ushort_t)(u >> 16));
        }
        d *= 0.125f;                             // 1/sqrt(64)
        Ss[b][c] = d;
        mx = fmaxf(mx, d);
    }
    float sum = 0.f;
    for (int c = 0; c < 32; c++) {
        float e = __expf(Ss[b][c] - mx);
        Ss[b][c] = e;
        sum += e;
    }
    float inv = 1.0f / sum;

    float oacc[HD];
#pragma unroll
    for (int d = 0; d < HD; d++) oacc[d] = 0.f;
    for (int c = 0; c < 32; c++) {
        float w = Ss[b][c] * inv;
        const uint_t* vr = (const uint_t*)&Vh[c * HD];
#pragma unroll
        for (int j = 0; j < HD / 2; j++) {
            uint_t u = vr[j];                    // broadcast LDS read
            oacc[2 * j]     += w * bf2f((ushort_t)u);
            oacc[2 * j + 1] += w * bf2f((ushort_t)(u >> 16));
        }
    }
    ushort_t* op = o + base;
#pragma unroll
    for (int d = 0; d < HD; d++) op[d] = f2bf(oacc[d]);
}

// ---------------- orchestration ----------------
extern "C" void kernel_launch(void* const* d_in, const int* in_sizes, int n_in,
                              void* d_out, int out_size, void* d_ws, size_t ws_size,
                              hipStream_t stream) {
    const float* x    = (const float*)d_in[0];
    const float* dw1w = (const float*)d_in[1];
    const float* dw1b = (const float*)d_in[2];
    const float* pw1w = (const float*)d_in[3];
    const float* pw1b = (const float*)d_in[4];
    const float* dw2w = (const float*)d_in[5];
    const float* dw2b = (const float*)d_in[6];
    const float* pw2w = (const float*)d_in[7];
    const float* pw2b = (const float*)d_in[8];
    const float* wq   = (const float*)d_in[9];
    const float* bq   = (const float*)d_in[10];
    const float* wk   = (const float*)d_in[11];
    const float* bk   = (const float*)d_in[12];
    const float* wv   = (const float*)d_in[13];
    const float* bv   = (const float*)d_in[14];
    const float* wp   = (const float*)d_in[15];
    const float* bp   = (const float*)d_in[16];
    const float* ffw  = (const float*)d_in[17];
    const float* ffb  = (const float*)d_in[18];

    char* ws = (char*)d_ws;
    size_t cur = 0;
    auto carve = [&](size_t bytes) -> void* {
        void* p = ws + cur;
        cur += (bytes + 255) & ~(size_t)255;
        return p;
    };
    const size_t NEL = (size_t)Mtot * Fv;
    float*    bufx = (float*)carve(NEL * 4);
    ushort_t* bfA  = (ushort_t*)carve(NEL * 2);
    ushort_t* bfB  = (ushort_t*)carve(NEL * 2);
    ushort_t* qb   = (ushort_t*)carve(NEL * 2);
    ushort_t* kb   = (ushort_t*)carve(NEL * 2);
    ushort_t* vb   = (ushort_t*)carve(NEL * 2);
    ushort_t* ab   = (ushort_t*)carve(NEL * 2);
    const int WN = Fv * Fv;
    ushort_t* pw1w_bf = (ushort_t*)carve((size_t)WN * 2);
    ushort_t* pw2w_bf = (ushort_t*)carve((size_t)WN * 2);
    ushort_t* wq_bf   = (ushort_t*)carve((size_t)WN * 2);
    ushort_t* wk_bf   = (ushort_t*)carve((size_t)WN * 2);
    ushort_t* wv_bf   = (ushort_t*)carve((size_t)WN * 2);
    ushort_t* wp_bf   = (ushort_t*)carve((size_t)WN * 2);
    ushort_t* ffw_bf  = (ushort_t*)carve((size_t)WN * 2);
    float*    stats   = (float*)carve((size_t)2 * Tv * 4);
    float*    part    = (float*)carve((size_t)2 * Bv * PPART * 4);

    const float INVN = 1.0f / (float)(Tv * Fv);
    const int EW = 1024;
    k_f2bf<<<EW, 256, 0, stream>>>(pw1w, pw1w_bf, WN);
    k_f2bf<<<EW, 256, 0, stream>>>(pw2w, pw2w_bf, WN);
    k_f2bf<<<EW, 256, 0, stream>>>(wq,   wq_bf,   WN);
    k_f2bf<<<EW, 256, 0, stream>>>(wk,   wk_bf,   WN);
    k_f2bf<<<EW, 256, 0, stream>>>(wv,   wv_bf,   WN);
    k_f2bf<<<EW, 256, 0, stream>>>(wp,   wp_bf,   WN);
    k_f2bf<<<EW, 256, 0, stream>>>(ffw,  ffw_bf,  WN);

    const int GE = (int)(NEL / 256);
    dim3 gemm_grid(Fv / 64, Mtot / 128);
    dim3 part_grid(PPART, Bv);

    // LN(x + pe) fused into depthwise conv 1; pointwise = WMMA GEMM
    k_stats_pe_part<<<part_grid, 256, 0, stream>>>(x, part);
    k_stats_fin<<<Bv, 32, 0, stream>>>(part, stats, INVN);
    k_ln_dwconv_pe<<<GE, 256, 0, stream>>>(x, stats, dw1w, dw1b, bfB);
    k_gemm<0, 0><<<gemm_grid, 256, 0, stream>>>(bfB, pw1w_bf, pw1b, nullptr, bufx, nullptr);

    // 3x residual conv blocks (shared weights): x = x + pw2(dw2(ln2(x)))
    for (int it = 0; it < 3; it++) {
        k_stats_b_part<<<part_grid, 256, 0, stream>>>(bufx, part);
        k_stats_fin<<<Bv, 32, 0, stream>>>(part, stats, INVN);
        k_ln_dwconv<<<GE, 256, 0, stream>>>(bufx, stats, dw2w, dw2b, bfB);
        k_gemm<1, 0><<<gemm_grid, 256, 0, stream>>>(bfB, pw2w_bf, pw2b, bufx, bufx, nullptr);
    }

    // y = ln2(x) (per-b); q/k/v projections (WMMA, bf16 out)
    k_stats_b_part<<<part_grid, 256, 0, stream>>>(bufx, part);
    k_stats_fin<<<Bv, 32, 0, stream>>>(part, stats, INVN);
    k_norm_b<<<GE, 256, 0, stream>>>(bufx, stats, bfA);
    k_gemm<0, 1><<<gemm_grid, 256, 0, stream>>>(bfA, wq_bf, bq, nullptr, nullptr, qb);
    k_gemm<0, 1><<<gemm_grid, 256, 0, stream>>>(bfA, wk_bf, bk, nullptr, nullptr, kb);
    k_gemm<0, 1><<<gemm_grid, 256, 0, stream>>>(bfA, wv_bf, bv, nullptr, nullptr, vb);

    // batch-axis attention per (t, h)
    dim3 attn_grid(Tv, Hh);
    k_attn<<<attn_grid, 32, 0, stream>>>(qb, kb, vb, ab);

    // x = x + proj(attn)
    k_gemm<1, 0><<<gemm_grid, 256, 0, stream>>>(ab, wp_bf, bp, bufx, bufx, nullptr);

    // per-t LN over (B,F); x = x + y @ ff_w.T + ff_b -> fp32 straight into d_out
    k_stats_t<<<Tv, 256, 0, stream>>>(bufx, stats);
    k_norm_t<<<GE, 256, 0, stream>>>(bufx, stats, bfA);
    k_gemm<1, 0><<<gemm_grid, 256, 0, stream>>>(bfA, ffw_bf, ffb, bufx, (float*)d_out, nullptr);
}
